// CausalTemporalAttention_77988016161248
// MI455X (gfx1250) — matrix operand order repaired and checked by reading, not compiled, not run on hardware
//
#include <hip/hip_runtime.h>

typedef __bf16 bf16_t;
typedef __attribute__((ext_vector_type(16))) __bf16 v16bf;
typedef __attribute__((ext_vector_type(8)))  __bf16 v8bf;
typedef __attribute__((ext_vector_type(8)))  float  v8f;

#define LN 2048
#define DN 1024
#define HN 16
#define HD 64

__device__ __forceinline__ v8bf ld8(const bf16_t* p) { return *(const v8bf*)p; }

__device__ __forceinline__ v16bf cat8(v8bf lo, v8bf hi) {
  return __builtin_shufflevector(lo, hi, 0,1,2,3,4,5,6,7,8,9,10,11,12,13,14,15);
}

// ---------------- fp32 -> bf16 conversion ----------------
__global__ void cvt_bf16_kernel(const float* __restrict__ s, bf16_t* __restrict__ d, int n) {
  int i = blockIdx.x * blockDim.x + threadIdx.x;
  if (i < n) d[i] = (bf16_t)s[i];
}

// ---------------- 64x64-per-wave bf16 WMMA GEMM: y = A @ W^T + bias --------
// MODE 0: q -> (B,H,L,HD) bf16   MODE 1: k -> (B,H,L,HD) bf16
// MODE 2: v -> (B,H,HD,L) bf16   MODE 3: fp32 row-major (final projection)
// 2 waves/SIMD min occupancy: 16 independent WMMA chains per wave already
// hide latency; a bigger VGPR budget (128-reg accumulator) beats occupancy.
template <int MODE>
__global__ void __launch_bounds__(256, 2)
gemm64_kernel(const bf16_t* __restrict__ A,   // (4096,1024)
              const bf16_t* __restrict__ W,   // (1024,1024) W[n][k]
              const float*  __restrict__ bias,
              bf16_t* __restrict__ outb,
              float*  __restrict__ outf)
{
  const int lane = threadIdx.x & 31;
  const int widx = threadIdx.x >> 5;
  const int wid  = blockIdx.x * 8 + widx;   // 1024 waves total
  const int m0 = (wid >> 4) * 64;           // 64 M tiles
  const int n0 = (wid & 15) * 64;           // 16 N tiles
  const int half = lane >> 4, lr = lane & 15;

  const bf16_t* a0 = A + (size_t)(m0 + lr) * DN;   // +i*16*DN via imm offsets
  const bf16_t* b0 = W + (size_t)(n0 + lr) * DN;   // +j*16*DN via imm offsets

  v8f acc[4][4] = {};
  for (int k0 = 0; k0 < DN; k0 += 32) {
    __builtin_prefetch(a0 + k0 + 128, 0, 1);
    v16bf af[4], bf[4];
#pragma unroll
    for (int i = 0; i < 4; i++)
      af[i] = cat8(ld8(a0 + i * 16 * DN + k0 + half * 8),
                   ld8(a0 + i * 16 * DN + k0 + 16 + half * 8));
#pragma unroll
    for (int j = 0; j < 4; j++)
      bf[j] = cat8(ld8(b0 + j * 16 * DN + k0 + half * 8),
                   ld8(b0 + j * 16 * DN + k0 + 16 + half * 8));
#pragma unroll
    for (int i = 0; i < 4; i++)
#pragma unroll
      for (int j = 0; j < 4; j++)
        acc[i][j] = __builtin_amdgcn_wmma_f32_16x16x32_bf16(
            false, af[i], false, bf[j], (short)0, acc[i][j], false, false);
  }

#pragma unroll
  for (int j = 0; j < 4; j++) {
    const int n = n0 + 16 * j + lr;
    const float bv = bias[n];
    const int h = n >> 6, d = n & 63;
#pragma unroll
    for (int i = 0; i < 4; i++) {
#pragma unroll
      for (int r = 0; r < 8; r++) {
        const int gm = m0 + 16 * i + half * 8 + r;   // 0..4095
        const float val = acc[i][j][r] + bv;
        if (MODE == 0 || MODE == 1) {
          const int b = gm >> 11, l = gm & 2047;
          outb[((size_t)(b * HN + h) * LN + l) * HD + d] = (bf16_t)val;
        } else if (MODE == 2) {
          const int b = gm >> 11, l = gm & 2047;
          outb[((size_t)(b * HN + h) * HD + d) * LN + l] = (bf16_t)val;
        } else {
          outf[(size_t)gm * DN + n] = val;
        }
      }
    }
  }
}

// ---------------- flash attention: causal + temporal decay ----------------
// one wave per (b, h, 32-query tile). Lane i owns softmax row m0+i:
// score tiles go C-layout -> LDS(f32) -> per-lane row softmax (pure VALU,
// no cross-lane permutes) -> row-major bf16 P (= A-fragment layout) + a
// 32-float broadcast of rescale factors back to the C-layout accumulators.
__global__ void __launch_bounds__(256, 2)
flash_attn_kernel(const bf16_t* __restrict__ qm,
                  const bf16_t* __restrict__ km,
                  const bf16_t* __restrict__ vtm,
                  const int*    __restrict__ pmask,
                  bf16_t* __restrict__ ao) // (B,L,D) bf16
{
  __shared__ float  slds[8][32][36];   // padded f32 scores, 36.9 KB
  __shared__ bf16_t plds[8][32][48];   // padded bf16 P rows, 24.6 KB
  __shared__ float  bbuf[8][32];       // per-row broadcast, 1 KB

  const int lane = threadIdx.x & 31;
  const int widx = threadIdx.x >> 5;
  const int wid = blockIdx.x * 8 + widx;
  const int qt = wid & 63;             // 64 query tiles of 32 rows
  const int bh = wid >> 6;             // 0..31
  const int b = bh >> 4, h = bh & 15;
  const int m0 = qt * 32;
  const int half = lane >> 4, lr = lane & 15;
  const float scale = 0.125f;          // hd^-0.5
  const float NEG = -1e30f;

  // Q fragments: [row-block g][hd chunk]
  const bf16_t* qbase = qm + ((size_t)bh * LN + m0 + lr) * HD;
  v16bf aq[2][2];
#pragma unroll
  for (int g = 0; g < 2; g++) {
    const bf16_t* qb = qbase + g * 16 * HD;
    aq[g][0] = cat8(ld8(qb +      half * 8), ld8(qb + 16 + half * 8));
    aq[g][1] = cat8(ld8(qb + 32 + half * 8), ld8(qb + 48 + half * 8));
  }

  v8f acc[2][4] = {};
  float m_row = NEG, s_row = 0.f;      // this lane's row statistics

  const int numkt = (m0 >> 5) + 1;     // 32-key tiles needed (causal)
  for (int kt = 0; kt < numkt; kt++) {
    const int n0 = kt * 32;
    __builtin_prefetch(km + ((size_t)bh * LN + n0 + 32 + lr) * HD, 0, 1);

    // K fragments, shared across both query row-blocks
    v16bf bk[2][2];
#pragma unroll
    for (int c = 0; c < 2; c++) {
      const bf16_t* krow = km + ((size_t)bh * LN + n0 + c * 16 + lr) * HD;
      bk[c][0] = cat8(ld8(krow +      half * 8), ld8(krow + 16 + half * 8));
      bk[c][1] = cat8(ld8(krow + 32 + half * 8), ld8(krow + 48 + half * 8));
    }

    const int col0 = n0 + lr, col1 = n0 + 16 + lr;
    const int pm0 = pmask[b * LN + col0], pm1 = pmask[b * LN + col1];

    // scores in C-layout: mask + decay, then park in LDS
#pragma unroll
    for (int g = 0; g < 2; g++) {
      v8f s0 = {}, s1 = {};
      s0 = __builtin_amdgcn_wmma_f32_16x16x32_bf16(false, aq[g][0], false, bk[0][0], (short)0, s0, false, false);
      s0 = __builtin_amdgcn_wmma_f32_16x16x32_bf16(false, aq[g][1], false, bk[0][1], (short)0, s0, false, false);
      s1 = __builtin_amdgcn_wmma_f32_16x16x32_bf16(false, aq[g][0], false, bk[1][0], (short)0, s1, false, false);
      s1 = __builtin_amdgcn_wmma_f32_16x16x32_bf16(false, aq[g][1], false, bk[1][1], (short)0, s1, false, false);
#pragma unroll
      for (int r = 0; r < 8; r++) {
        const int row = m0 + g * 16 + half * 8 + r;
        float x0 = s0[r] * scale - 0.1f * (float)(row - col0);
        float x1 = s1[r] * scale - 0.1f * (float)(row - col1);
        x0 = (col0 <= row && pm0 != 0) ? x0 : NEG;
        x1 = (col1 <= row && pm1 != 0) ? x1 : NEG;
        slds[widx][g * 16 + half * 8 + r][lr]      = x0;
        slds[widx][g * 16 + half * 8 + r][16 + lr] = x1;
      }
    }
    __builtin_amdgcn_wave_barrier();
    asm volatile("s_wait_dscnt 0" ::: "memory");

    // per-lane row softmax (lane owns row m0+lane)
    float xr[32];
#pragma unroll
    for (int j = 0; j < 32; j++) xr[j] = slds[widx][lane][j];
    float tmax = NEG;
#pragma unroll
    for (int j = 0; j < 32; j++) tmax = fmaxf(tmax, xr[j]);
    const float nm = fmaxf(m_row, tmax);
    const float rescale = __expf(m_row - nm);
    float rs = 0.f;
#pragma unroll
    for (int j = 0; j < 32; j++) { xr[j] = __expf(xr[j] - nm); rs += xr[j]; }
    s_row = s_row * rescale + rs;
    m_row = nm;

    // write bf16 P row (row-major == A-fragment layout) + rescale broadcast
#pragma unroll
    for (int cch = 0; cch < 4; cch++) {
      v8bf pk;
#pragma unroll
      for (int e = 0; e < 8; e++) pk[e] = (bf16_t)xr[cch * 8 + e];
      *(v8bf*)&plds[widx][lane][cch * 8] = pk;
    }
    bbuf[widx][lane] = rescale;
    __builtin_amdgcn_wave_barrier();
    asm volatile("s_wait_dscnt 0" ::: "memory");

    // rescale C-layout accumulators with broadcast factors
#pragma unroll
    for (int g = 0; g < 2; g++) {
      const float4 fa = *(const float4*)&bbuf[widx][g * 16 + half * 8];
      const float4 fb = *(const float4*)&bbuf[widx][g * 16 + half * 8 + 4];
      const float rf[8] = {fa.x, fa.y, fa.z, fa.w, fb.x, fb.y, fb.z, fb.w};
#pragma unroll
      for (int r = 0; r < 8; r++) {
        acc[g][0][r] *= rf[r]; acc[g][1][r] *= rf[r];
        acc[g][2][r] *= rf[r]; acc[g][3][r] *= rf[r];
      }
    }

    // P tiles as A fragments (per row-block)
    v16bf ap[2];
#pragma unroll
    for (int g = 0; g < 2; g++)
      ap[g] = cat8(ld8(&plds[widx][g * 16 + lr][half * 8]),
                   ld8(&plds[widx][g * 16 + lr][16 + half * 8]));

    // O += P @ V, V fragments shared across row-blocks
#pragma unroll
    for (int t = 0; t < 4; t++) {
      const bf16_t* vr = vtm + (size_t)(bh * HD + t * 16 + lr) * LN + n0 + half * 8;
      v16bf bv = cat8(ld8(vr), ld8(vr + 16));
      acc[0][t] = __builtin_amdgcn_wmma_f32_16x16x32_bf16(false, ap[0], false, bv, (short)0, acc[0][t], false, false);
      acc[1][t] = __builtin_amdgcn_wmma_f32_16x16x32_bf16(false, ap[1], false, bv, (short)0, acc[1][t], false, false);
    }
  }

  // broadcast 1/s per row, normalize, store (B,L,D) bf16
  bbuf[widx][lane] = s_row > 0.f ? 1.f / s_row : 0.f;
  __builtin_amdgcn_wave_barrier();
  asm volatile("s_wait_dscnt 0" ::: "memory");
#pragma unroll
  for (int g = 0; g < 2; g++) {
    const float4 fa = *(const float4*)&bbuf[widx][g * 16 + half * 8];
    const float4 fb = *(const float4*)&bbuf[widx][g * 16 + half * 8 + 4];
    const float rf[8] = {fa.x, fa.y, fa.z, fa.w, fb.x, fb.y, fb.z, fb.w};
#pragma unroll
    for (int r = 0; r < 8; r++) {
      const int l = m0 + g * 16 + half * 8 + r;
      bf16_t* op = ao + ((size_t)b * LN + l) * DN + h * HD + lr;
      op[0]  = (bf16_t)(acc[g][0][r] * rf[r]);
      op[16] = (bf16_t)(acc[g][1][r] * rf[r]);
      op[32] = (bf16_t)(acc[g][2][r] * rf[r]);
      op[48] = (bf16_t)(acc[g][3][r] * rf[r]);
    }
  }
}

extern "C" void kernel_launch(void* const* d_in, const int* in_sizes, int n_in,
                              void* d_out, int out_size, void* d_ws, size_t ws_size,
                              hipStream_t stream) {
  const float* x  = (const float*)d_in[0];
  const float* wq = (const float*)d_in[1];
  const float* bq = (const float*)d_in[2];
  const float* wk = (const float*)d_in[3];
  const float* bk = (const float*)d_in[4];
  const float* wv = (const float*)d_in[5];
  const float* bv = (const float*)d_in[6];
  const float* wo = (const float*)d_in[7];
  const float* bo = (const float*)d_in[8];
  const int*   pm = (const int*)d_in[9];
  float* out = (float*)d_out;

  const size_t XB = (size_t)4096 * 1024;   // B*L*D elements
  const size_t WB = (size_t)1024 * 1024;

  bf16_t* xb  = (bf16_t*)d_ws;             // 48 MB total workspace use
  bf16_t* wqb = xb  + XB;
  bf16_t* wkb = wqb + WB;
  bf16_t* wvb = wkb + WB;
  bf16_t* wob = wvb + WB;
  bf16_t* qm  = wob + WB;
  bf16_t* km  = qm  + XB;
  bf16_t* vtm = km  + XB;
  bf16_t* ab  = vtm + XB;

  cvt_bf16_kernel<<<(int)((XB + 255) / 256), 256, 0, stream>>>(x,  xb,  (int)XB);
  cvt_bf16_kernel<<<(int)((WB + 255) / 256), 256, 0, stream>>>(wq, wqb, (int)WB);
  cvt_bf16_kernel<<<(int)((WB + 255) / 256), 256, 0, stream>>>(wk, wkb, (int)WB);
  cvt_bf16_kernel<<<(int)((WB + 255) / 256), 256, 0, stream>>>(wv, wvb, (int)WB);
  cvt_bf16_kernel<<<(int)((WB + 255) / 256), 256, 0, stream>>>(wo, wob, (int)WB);

  // 64 x 16 = 1024 waves, 8 per block -> 128 blocks per matrix
  gemm64_kernel<0><<<128, 256, 0, stream>>>(xb, wqb, bq, qm,  nullptr);
  gemm64_kernel<1><<<128, 256, 0, stream>>>(xb, wkb, bk, km,  nullptr);
  gemm64_kernel<2><<<128, 256, 0, stream>>>(xb, wvb, bv, vtm, nullptr);

  // 2*16*64 = 2048 waves, 8 per block -> 256 blocks
  flash_attn_kernel<<<256, 256, 0, stream>>>(qm, km, vtm, pm, ab);

  gemm64_kernel<3><<<128, 256, 0, stream>>>(ab, wob, bo, nullptr, out);
}
